// Model_987842478152
// MI455X (gfx1250) — matrix operand order repaired and checked by reading, not compiled, not run on hardware
//
#include <hip/hip_runtime.h>
#include <hip/hip_bf16.h>

// ---------------------------------------------------------------------------
// Model constants (match reference)
// ---------------------------------------------------------------------------
constexpr int kB    = 512;
constexpr int kT    = 64;
constexpr int kIN   = 16;
constexpr int kHID  = 128;
constexpr int kHEAD = 4;
constexpr int kDF   = 32;
constexpr int kBT   = kB * kT;              // 32768 rows
constexpr int kEBT  = kBT * kHID;           // 4,194,304 floats per slab
constexpr int kPAD  = kB * (kT + 7) * kHID; // worst-case padded slab (w=8)
constexpr float kInvSqrtDF = 0.17677669529663687f; // 1/sqrt(32)

typedef __attribute__((ext_vector_type(4)))  float    f32x4;
typedef __attribute__((ext_vector_type(2)))  _Float16 v2h;
typedef __attribute__((ext_vector_type(4)))  _Float16 v4h;
typedef __attribute__((ext_vector_type(8)))  _Float16 v8h;
typedef __attribute__((ext_vector_type(16))) _Float16 v16h;
typedef __attribute__((ext_vector_type(8)))  float    v8f;

// ---------------------------------------------------------------------------
// Tiled WMMA GEMM:  C[M,N] = act(A[M,K] @ B[K,N] + bias[N])
// REQUIRES: M % 128 == 0, N % 64 == 0, K % 32 == 0 (caller guarantees; the
// K=16 input linears are zero-padded to K=32 first).
// Block = 256 threads (8 waves), block tile 128x64. Each wave computes a
// 32x32 output patch: 2 A fragments x 2 B fragments -> 4 v_wmma per K-step,
// f32 accumulation. A staged row-major f16 (stride 40 to kill bank
// conflicts); B staged TRANSPOSED (Bs[n][k]) so both fragment types are two
// contiguous ds_load_b128 per lane, matching the ISA 16-bit A(16x32) /
// B(32x16) VGPR layouts exactly.
// ---------------------------------------------------------------------------
__global__ __launch_bounds__(256) void k_gemm(const float* __restrict__ A,
                                              const float* __restrict__ Bm,
                                              const float* __restrict__ bias,
                                              float* __restrict__ C,
                                              int M, int N, int K, int act) {
  __shared__ __attribute__((aligned(16))) _Float16 As[128 * 40]; // 10 KB
  __shared__ __attribute__((aligned(16))) _Float16 Bs[64 * 40];  //  5 KB

  const int tid   = threadIdx.x;
  const int tileM = blockIdx.y * 128;
  const int tileN = blockIdx.x * 64;
  const int wave  = tid >> 5;
  const int lane  = tid & 31;
  const int wy    = wave & 3;   // row group (of 32)
  const int wx    = wave >> 2;  // col group (of 32)
  const int m16   = lane & 15;
  const int lh    = lane >> 4;  // lane half (0/1)

  v8f c00 = {}, c01 = {}, c10 = {}, c11 = {};

  // per-thread staging coordinates (constant over K loop)
  const int aRow = tid >> 3;            // p = tid+256u -> row = p>>3
  const int aK4  = (tid & 7) * 4;       // float4 column within tile
  const int bN   = tid & 63;            // p = tid+256u -> n = p&63
  const int bK0  = (tid >> 6) * 2;      // k pair base (increments by 8 per u)

  for (int kk = 0; kk < K; kk += 32) {
    // ---- gather global tiles into registers (no LDS dependency yet) ----
    f32x4 aReg[4];
    float bR0[4], bR1[4];
#pragma unroll
    for (int u = 0; u < 4; ++u) {
      int row = aRow + u * 32;                       // (tid+256u)>>3
      aReg[u] = *(const f32x4*)(A + (size_t)(tileM + row) * K + kk + aK4);
    }
#pragma unroll
    for (int u = 0; u < 4; ++u) {
      int k = bK0 + u * 8;                           // ((tid+256u)>>6)*2
      const float* bp = Bm + (size_t)(kk + k) * N + tileN + bN;
      bR0[u] = bp[0];
      bR1[u] = bp[N];
    }
    if (kk + 32 < K) {
      __builtin_prefetch(A + (size_t)(tileM + (tid & 127)) * K + kk + 32, 0, 1);
      __builtin_prefetch(Bm + (size_t)(kk + 32 + (tid & 31)) * N + tileN, 0, 1);
    }

    __syncthreads();   // previous iteration's fragment reads are done

    // ---- convert f32 -> f16 and store to LDS ----
#pragma unroll
    for (int u = 0; u < 4; ++u) {
      int row = aRow + u * 32;
      v4h pa = __builtin_convertvector(aReg[u], v4h);
      *(v4h*)(&As[row * 40 + aK4]) = pa;
    }
#pragma unroll
    for (int u = 0; u < 4; ++u) {
      int k = bK0 + u * 8;
      v2h pb = {(_Float16)bR0[u], (_Float16)bR1[u]};
      *(v2h*)(&Bs[bN * 40 + k]) = pb;                // transposed: Bs[n][k]
    }
    __syncthreads();

    // ---- fragment loads: two contiguous b128 per operand per lane ----
    auto ldA = [&](int row) -> v16h {
      const _Float16* base = &As[row * 40];
      v8h lo = *(const v8h*)(base + lh * 8);        // K = lh*8 + 0..7
      v8h hi = *(const v8h*)(base + 16 + lh * 8);   // K = 16 + lh*8 + 0..7
      return __builtin_shufflevector(lo, hi, 0, 1, 2, 3, 4, 5, 6, 7,
                                     8, 9, 10, 11, 12, 13, 14, 15);
    };
    auto ldB = [&](int n) -> v16h {
      const _Float16* base = &Bs[n * 40];
      v8h lo = *(const v8h*)(base + lh * 8);        // K = lh*8 + 0..7
      v8h hi = *(const v8h*)(base + 16 + lh * 8);   // K = 16 + lh*8 + 0..7
      return __builtin_shufflevector(lo, hi, 0, 1, 2, 3, 4, 5, 6, 7,
                                     8, 9, 10, 11, 12, 13, 14, 15);
    };
    v16h a0 = ldA(wy * 32 + m16);
    v16h a1 = ldA(wy * 32 + 16 + m16);
    v16h b0 = ldB(wx * 32 + m16);
    v16h b1 = ldB(wx * 32 + 16 + m16);

    c00 = __builtin_amdgcn_wmma_f32_16x16x32_f16(false, a0, false, b0,
                                                 (short)0, c00, false, false);
    c01 = __builtin_amdgcn_wmma_f32_16x16x32_f16(false, a0, false, b1,
                                                 (short)0, c01, false, false);
    c10 = __builtin_amdgcn_wmma_f32_16x16x32_f16(false, a1, false, b0,
                                                 (short)0, c10, false, false);
    c11 = __builtin_amdgcn_wmma_f32_16x16x32_f16(false, a1, false, b1,
                                                 (short)0, c11, false, false);
  }

  // ---- epilogue: C/D layout VGPR r -> row = lh*8 + r, col = lane%16 ----
  const int col0 = tileN + wx * 32 + m16;
  const int col1 = col0 + 16;
  float bia0 = 0.0f, bia1 = 0.0f;
  if (bias) { bia0 = bias[col0]; bia1 = bias[col1]; }
#pragma unroll
  for (int r = 0; r < 8; ++r) {
    int row0 = tileM + wy * 32 + lh * 8 + r;
    int row1 = row0 + 16;
    float v00 = c00[r] + bia0, v01 = c01[r] + bia1;
    float v10 = c10[r] + bia0, v11 = c11[r] + bia1;
    if (act == 1) {
      v00 = fmaxf(v00, 0.0f); v01 = fmaxf(v01, 0.0f);
      v10 = fmaxf(v10, 0.0f); v11 = fmaxf(v11, 0.0f);
    }
    C[(size_t)row0 * N + col0] = v00;
    C[(size_t)row0 * N + col1] = v01;
    C[(size_t)row1 * N + col0] = v10;
    C[(size_t)row1 * N + col1] = v11;
  }
}

// ---------------------------------------------------------------------------
// K-padding helpers so every GEMM sees K % 32 == 0.
// ---------------------------------------------------------------------------
// x [rows,16] -> xp [rows,32] zero-padded
__global__ void k_padx(const float* __restrict__ X, float* __restrict__ Xp,
                       int n) {
  int i = blockIdx.x * blockDim.x + threadIdx.x;
  if (i >= n) return;
  int c = i & 31;
  int r = i >> 5;
  Xp[i] = (c < kIN) ? X[(size_t)r * kIN + c] : 0.0f;
}
// W [16,128] -> Wp [32,128] zero-padded
__global__ void k_padw(const float* __restrict__ W, float* __restrict__ Wp,
                       int n) {
  int i = blockIdx.x * blockDim.x + threadIdx.x;
  if (i >= n) return;
  int r = i >> 7;
  Wp[i] = (r < kIN) ? W[i] : 0.0f;
}

// ---------------------------------------------------------------------------
// GRU gate:  acc = sigmoid(acc + other + bias[c])
// ---------------------------------------------------------------------------
__global__ void k_rz(float* __restrict__ acc, const float* __restrict__ other,
                     const float* __restrict__ bias, int n) {
  int i = blockIdx.x * blockDim.x + threadIdx.x;
  if (i >= n) return;
  float v = acc[i] + other[i] + bias[i & (kHID - 1)];
  acc[i] = 1.0f / (1.0f + __expf(-v));
}

__global__ void k_mul(float* __restrict__ dst, const float* __restrict__ a,
                      const float* __restrict__ b, int n) {
  int i = blockIdx.x * blockDim.x + threadIdx.x;
  if (i >= n) return;
  dst[i] = a[i] * b[i];
}

// h' = (1-z)*h + z*tanh(hh1 + hh2 + bh[c])
__global__ void k_newh(float* __restrict__ h, const float* __restrict__ z,
                       const float* __restrict__ hh1,
                       const float* __restrict__ hh2,
                       const float* __restrict__ bh, int n) {
  int i = blockIdx.x * blockDim.x + threadIdx.x;
  if (i >= n) return;
  float hhat = tanhf(hh1[i] + hh2[i] + bh[i & (kHID - 1)]);
  float zz = z[i];
  h[i] = (1.0f - zz) * h[i] + zz * hhat;
}

// zero-pad in time:  dst[b, j, c] = (j < w-1) ? 0 : src[b, j-(w-1), c]
__global__ void k_pad(float* __restrict__ dst, const float* __restrict__ src,
                      int w, int n) {
  int i = blockIdx.x * blockDim.x + threadIdx.x;
  if (i >= n) return;
  const int TP = kT + w - 1;
  int c = i & (kHID - 1);
  int j = (i >> 7) % TP;
  int b = (i >> 7) / TP;
  float v = 0.0f;
  if (j >= w - 1) v = src[((size_t)b * kT + (j - (w - 1))) * kHID + c];
  dst[i] = v;
}

// ---------------------------------------------------------------------------
// Causal sliding-window attention. Replicates the raw
// [B,T,w,128] -> [B,T,HEAD,w,DF] reshape: flat = h*w*32 + j*32 + d,
// source = (t + flat/128, flat%128) in the padded k/v tensors.
// One thread per (b,t,h).
// ---------------------------------------------------------------------------
__global__ void k_local(const float* __restrict__ Q, const float* __restrict__ KF,
                        const float* __restrict__ VF, float* __restrict__ O,
                        int w) {
  int idx = blockIdx.x * blockDim.x + threadIdx.x;
  if (idx >= kB * kT * kHEAD) return;
  int h = idx & 3;
  int t = (idx >> 2) & (kT - 1);
  int b = idx >> 8;
  const int TP = kT + w - 1;

  float q[kDF];
  const float* qp = Q + ((size_t)(b * kT + t)) * kHID + h * kDF;
#pragma unroll
  for (int d = 0; d < kDF; ++d) q[d] = qp[d];

  float s[8];
  float mx = -1e30f;
  for (int j = 0; j < w; ++j) {
    float acc = 0.0f;
#pragma unroll
    for (int d = 0; d < kDF; ++d) {
      int flat = (h * w + j) * kDF + d;
      int js = flat >> 7, cs = flat & 127;
      acc += q[d] * KF[((size_t)(b * TP + t + js)) * kHID + cs];
    }
    s[j] = acc * kInvSqrtDF;
    mx = fmaxf(mx, s[j]);
  }
  float sum = 0.0f;
  for (int j = 0; j < w; ++j) {
    s[j] = __expf(s[j] - mx);
    sum += s[j];
  }
  float inv = 1.0f / sum;

  float o[kDF];
#pragma unroll
  for (int d = 0; d < kDF; ++d) o[d] = 0.0f;
  for (int j = 0; j < w; ++j) {
    float aj = s[j] * inv;
#pragma unroll
    for (int d = 0; d < kDF; ++d) {
      int flat = (h * w + j) * kDF + d;
      int js = flat >> 7, cs = flat & 127;
      o[d] += aj * VF[((size_t)(b * TP + t + js)) * kHID + cs];
    }
  }
  float* op = O + ((size_t)(b * kT + t)) * kHID + h * kDF;
#pragma unroll
  for (int d = 0; d < kDF; ++d) op[d] = o[d];
}

// ---------------------------------------------------------------------------
// TAA: full T x T attention with z_hat distance gating.
// relu(softmax(.)) == softmax(.), so relu is dropped. Output written at the
// raw-reshape position f[b, h*T*DF + t*DF + d]. Streaming softmax.
// One thread per (b,h,t).
// ---------------------------------------------------------------------------
__global__ void k_taa(const float* __restrict__ Q, const float* __restrict__ Kt,
                      const float* __restrict__ Vt,
                      const float* __restrict__ ahp,
                      const float* __restrict__ vhp, float* __restrict__ F) {
  int idx = blockIdx.x * blockDim.x + threadIdx.x;
  if (idx >= kB * kHEAD * kT) return;
  int t = idx & (kT - 1);
  int h = (idx >> 6) & 3;
  int b = idx >> 8;
  const float ah = ahp[0];
  const float vh = vhp[0];
  const float evh = __expf(vh);

  float q[kDF];
  const float* qp = Q + ((size_t)(b * kT + t)) * kHID + h * kDF;
#pragma unroll
  for (int d = 0; d < kDF; ++d) q[d] = qp[d];

  float mx = -1e30f;
  for (int s = 0; s < kT; ++s) {
    const float* kp = Kt + ((size_t)(b * kT + s)) * kHID + h * kDF;
    float acc = 0.0f;
#pragma unroll
    for (int d = 0; d < kDF; ++d) acc += q[d] * kp[d];
    float dist = fabsf((float)(t - s));
    float zh = 1.0f + evh / (1.0f + __expf(vh - ah * dist));
    mx = fmaxf(mx, acc * zh * kInvSqrtDF);
  }
  float sum = 0.0f;
  float o[kDF];
#pragma unroll
  for (int d = 0; d < kDF; ++d) o[d] = 0.0f;
  for (int s = 0; s < kT; ++s) {
    const float* kp = Kt + ((size_t)(b * kT + s)) * kHID + h * kDF;
    float acc = 0.0f;
#pragma unroll
    for (int d = 0; d < kDF; ++d) acc += q[d] * kp[d];
    float dist = fabsf((float)(t - s));
    float zh = 1.0f + evh / (1.0f + __expf(vh - ah * dist));
    float wgt = __expf(acc * zh * kInvSqrtDF - mx);
    sum += wgt;
    const float* vp = Vt + ((size_t)(b * kT + s)) * kHID + h * kDF;
#pragma unroll
    for (int d = 0; d < kDF; ++d) o[d] += wgt * vp[d];
  }
  float inv = 1.0f / sum;
  float* fp = F + (size_t)b * kT * kHID + h * (kT * kDF) + t * kDF;
#pragma unroll
  for (int d = 0; d < kDF; ++d) fp[d] = o[d] * inv;
}

// Add + LayerNorm over last dim. One thread per row of 128.
__global__ void k_addln(const float* __restrict__ X, const float* __restrict__ Fin,
                        const float* __restrict__ g, const float* __restrict__ bb,
                        float* __restrict__ Y, int rows) {
  int r = blockIdx.x * blockDim.x + threadIdx.x;
  if (r >= rows) return;
  const float* x = X + (size_t)r * kHID;
  const float* f = Fin + (size_t)r * kHID;
  float mu = 0.0f;
  for (int c = 0; c < kHID; ++c) mu += x[c] + f[c];
  mu *= (1.0f / kHID);
  float var = 0.0f;
  for (int c = 0; c < kHID; ++c) {
    float v = x[c] + f[c] - mu;
    var += v * v;
  }
  var *= (1.0f / kHID);
  float inv = rsqrtf(var + 1e-5f);
  float* y = Y + (size_t)r * kHID;
  for (int c = 0; c < kHID; ++c)
    y[c] = (x[c] + f[c] - mu) * inv * g[c] + bb[c];
}

// mean over T -> output[b, i*HID + c]   (pooled.transpose(1,0,2).reshape)
__global__ void k_pool(const float* __restrict__ Y, float* __restrict__ Out,
                       int wsIdx) {
  int idx = blockIdx.x * blockDim.x + threadIdx.x;
  if (idx >= kB * kHID) return;
  int c = idx & (kHID - 1);
  int b = idx >> 7;
  float s = 0.0f;
  for (int t = 0; t < kT; ++t) s += Y[((size_t)b * kT + t) * kHID + c];
  Out[(size_t)b * (2 * kHID) + wsIdx * kHID + c] = s * (1.0f / kT);
}

// final head: y = LeakyReLU_{0.1}(output @ W[256,1] + b)
__global__ void k_head(const float* __restrict__ Omat, const float* __restrict__ W,
                       const float* __restrict__ bias, float* __restrict__ Ret) {
  int b = blockIdx.x * blockDim.x + threadIdx.x;
  if (b >= kB) return;
  float acc = bias[0];
  for (int j = 0; j < 2 * kHID; ++j) acc += Omat[(size_t)b * 2 * kHID + j] * W[j];
  Ret[b] = acc > 0.0f ? acc : 0.1f * acc;
}

// ---------------------------------------------------------------------------
// Host orchestration
// ---------------------------------------------------------------------------
struct GruP {
  const float *WrW, *WrB, *UrW, *UrB, *br;
  const float *WzW, *WzB, *UzW, *UzB, *bz;
  const float *WhW, *WhB, *UhW, *UhB, *bh;
};
struct LitP {
  const float *qW, *qB, *kW, *kB_, *vW, *vB;
  const float *ah, *vh, *tqW, *tqB, *tkW, *tkB, *tvW, *tvB;
  const float *l1W, *l1B, *l2W, *l2B;
};

extern "C" void kernel_launch(void* const* d_in, const int* in_sizes, int n_in,
                              void* d_out, int out_size, void* d_ws, size_t ws_size,
                              hipStream_t stream) {
  (void)in_sizes; (void)n_in; (void)out_size; (void)ws_size;

  // ----- unpack inputs: setup_inputs() dict insertion order, leaves in
  // recursive insertion order ({"W","b"} per linear, lists by index) -----
  int qi = 0;
  auto nxt = [&]() { return (const float*)d_in[qi++]; };
  const float* X   = nxt();   // [B,T,IN]
  const float* ADJ = nxt();   // [B,B]
  const float *embW = nxt(), *embB = nxt();
  const float *agg0W = nxt(), *agg0B = nxt();
  const float *agg1W = nxt(), *agg1B = nxt();
  GruP gp[2];
  for (int l = 0; l < 2; ++l) {
    gp[l].WrW = nxt(); gp[l].WrB = nxt();
    gp[l].UrW = nxt(); gp[l].UrB = nxt();
    gp[l].br  = nxt();
    gp[l].WzW = nxt(); gp[l].WzB = nxt();
    gp[l].UzW = nxt(); gp[l].UzB = nxt();
    gp[l].bz  = nxt();
    gp[l].WhW = nxt(); gp[l].WhB = nxt();
    gp[l].UhW = nxt(); gp[l].UhB = nxt();
    gp[l].bh  = nxt();
  }
  LitP lp[2];
  for (int i = 0; i < 2; ++i) {
    lp[i].qW = nxt(); lp[i].qB = nxt();
    lp[i].kW = nxt(); lp[i].kB_ = nxt();
    lp[i].vW = nxt(); lp[i].vB = nxt();
    lp[i].ah = nxt(); lp[i].vh = nxt();           // taa scalars
    lp[i].tqW = nxt(); lp[i].tqB = nxt();
    lp[i].tkW = nxt(); lp[i].tkB = nxt();
    lp[i].tvW = nxt(); lp[i].tvB = nxt();
    lp[i].l1W = nxt(); lp[i].l1B = nxt();
    lp[i].l2W = nxt(); lp[i].l2B = nxt();
  }
  const float *lnG = nxt(), *lnBi = nxt();
  const float *outW = nxt(), *outB = nxt();

  // ----- workspace slabs (floats) -----
  float* w    = (float*)d_ws;
  float* e    = w;             // hidden state
  float* a    = e + kEBT;      // aggregated
  float* t0   = a + kEBT;
  float* t1   = t0 + kEBT;
  float* t2   = t1 + kEBT;
  float* t3   = t2 + kEBT;
  float* outb = t3 + kEBT;     // local-attention output ("out" in reference)
  float* padb = outb + kEBT;   // zero-padded sequence (also x K-pad at start)
  float* kf   = padb + kPAD;
  float* vf   = kf + kPAD;
  float* fb   = vf + kPAD;     // taa output (raw-reshape layout)
  float* wp0  = fb + kEBT;     // agg0 W zero-padded to [32,128]
  float* wp1  = wp0 + 32 * kHID;  // emb W zero-padded to [32,128]

  float* dOut = (float*)d_out;               // [B, 2*HID] then [B] leaky y
  float* dRet = dOut + (size_t)kB * 2 * kHID;

  auto gemm = [&](const float* A, const float* Bm, const float* bias, float* C,
                  int M, int N, int K, int act) {
    dim3 g((unsigned)(N / 64), (unsigned)(M / 128));
    k_gemm<<<g, 256, 0, stream>>>(A, Bm, bias, C, M, N, K, act);
  };
  const int EW = (kEBT + 255) / 256;

  auto gruStep = [&](const GruP& g) {
    gemm(a, g.WrW, g.WrB, t0, kBT, kHID, kHID, 0);
    gemm(e, g.UrW, g.UrB, t1, kBT, kHID, kHID, 0);
    k_rz<<<EW, 256, 0, stream>>>(t0, t1, g.br, kEBT);            // r -> t0
    gemm(a, g.WzW, g.WzB, t1, kBT, kHID, kHID, 0);
    gemm(e, g.UzW, g.UzB, t2, kBT, kHID, kHID, 0);
    k_rz<<<EW, 256, 0, stream>>>(t1, t2, g.bz, kEBT);            // z -> t1
    k_mul<<<EW, 256, 0, stream>>>(t2, t0, e, kEBT);              // r*h -> t2
    gemm(t2, g.UhW, g.UhB, t3, kBT, kHID, kHID, 0);
    gemm(a, g.WhW, g.WhB, t2, kBT, kHID, kHID, 0);
    k_newh<<<EW, 256, 0, stream>>>(e, t1, t3, t2, g.bh, kEBT);   // e updated
  };

  // ----- K-pad the IN=16 operands so every GEMM has K % 32 == 0 -----
  float* xp = padb;  // [BT,32] fits in the pad slab (free until attention)
  k_padx<<<(kBT * 32 + 255) / 256, 256, 0, stream>>>(X, xp, kBT * 32);
  k_padw<<<(32 * kHID + 255) / 256, 256, 0, stream>>>(agg0W, wp0, 32 * kHID);
  k_padw<<<(32 * kHID + 255) / 256, 256, 0, stream>>>(embW, wp1, 32 * kHID);

  // ----- encoder: agg0 + emb + gru0, agg1 + gru1 -----
  gemm(xp, wp0, agg0B, t0, kBT, kHID, 32, 0);                    // lin(x)
  gemm(ADJ, t0, nullptr, a, kB, kT * kHID, kB, 0);               // adj @ .
  gemm(xp, wp1, embB, e, kBT, kHID, 32, 0);                      // emb
  gruStep(gp[0]);
  gemm(e, agg1W, agg1B, t0, kBT, kHID, kHID, 0);
  gemm(ADJ, t0, nullptr, a, kB, kT * kHID, kB, 0);
  gruStep(gp[1]);

  // ----- attention blocks -----
  const int WS[2] = {5, 8};
  const float* cur = e;
  for (int i = 0; i < 2; ++i) {
    const int wdw = WS[i];
    const int TP = kT + wdw - 1;
    const int nPad = kB * TP * kHID;

    k_pad<<<(nPad + 255) / 256, 256, 0, stream>>>(padb, cur, wdw, nPad);
    gemm(padb, lp[i].kW, lp[i].kB_, kf, kB * TP, kHID, kHID, 0);
    gemm(padb, lp[i].vW, lp[i].vB, vf, kB * TP, kHID, kHID, 0);
    gemm(cur, lp[i].qW, lp[i].qB, t0, kBT, kHID, kHID, 0);
    k_local<<<(kB * kT * kHEAD + 255) / 256, 256, 0, stream>>>(t0, kf, vf,
                                                               outb, wdw);
    // taa on outb
    gemm(outb, lp[i].tqW, lp[i].tqB, t0, kBT, kHID, kHID, 0);
    gemm(outb, lp[i].tkW, lp[i].tkB, t1, kBT, kHID, kHID, 0);
    gemm(outb, lp[i].tvW, lp[i].tvB, t2, kBT, kHID, kHID, 0);
    k_taa<<<(kB * kHEAD * kT + 255) / 256, 256, 0, stream>>>(t0, t1, t2,
                                                             lp[i].ah, lp[i].vh,
                                                             fb);
    // ffn
    gemm(fb, lp[i].l1W, lp[i].l1B, t0, kBT, kHID, kHID, 1);      // relu
    gemm(t0, lp[i].l2W, lp[i].l2B, t1, kBT, kHID, kHID, 0);
    // ln(out + ffn) -> t2 ; pool -> d_out
    k_addln<<<(kBT + 255) / 256, 256, 0, stream>>>(outb, t1, lnG, lnBi, t2, kBT);
    k_pool<<<(kB * kHID + 255) / 256, 256, 0, stream>>>(t2, dOut, i);

    cur = outb;  // next block consumes this block's local-attention output
  }

  // ----- head -----
  k_head<<<(kB + 255) / 256, 256, 0, stream>>>(dOut, outW, outB, dRet);
}
